// NonLocal_89206470738281
// MI455X (gfx1250) — compile-verified, matrix operand order checked by reading, not run on hardware
//
#include <hip/hip_runtime.h>
#include <hip/hip_bf16.h>

#define NB  32
#define CCH 256
#define DDH 128
#define HWS 1024
#define QT  32

typedef __bf16 bf16_t;
typedef __attribute__((ext_vector_type(16))) __bf16 v16bf;
typedef __attribute__((ext_vector_type(8)))  float  v8f;
typedef __attribute__((ext_vector_type(4)))  unsigned int v4u;
typedef __attribute__((ext_vector_type(8)))  int  v8i;
typedef __attribute__((ext_vector_type(4)))  int  v4i;

union AFragU { v16bf v; uint4 q[2]; };

// 16(M/N-rows) x 32(K) bf16 fragment from a row-major [rows x K] matrix (global or LDS).
// lane holds row (lane&15); K elements: base=(lane>>4)*8 -> [base..base+7] and [base+16..base+23]
__device__ __forceinline__ v16bf load_frag(const bf16_t* p, int ld, int lane) {
  const bf16_t* q = p + (long)(lane & 15) * ld + ((lane >> 4) << 3);
  AFragU f;
  f.q[0] = *(const uint4*)(q);
  f.q[1] = *(const uint4*)(q + 16);
  return f.v;
}

__device__ __forceinline__ v8f wmma_bf(v16bf a, v16bf b, v8f c) {
  return __builtin_amdgcn_wmma_f32_16x16x32_bf16(false, a, false, b, (short)0, c, false, false);
}

// C/D frag: lane holds col (lane&15); vgpr v holds row v + 8*(lane>>4)
__device__ __forceinline__ void store_cfrag_f32(float* p, int ld, int lane, v8f c, float alpha) {
  int col = lane & 15;
  int r0 = (lane >> 4) << 3;
#pragma unroll
  for (int v = 0; v < 8; ++v) p[(long)(r0 + v) * ld + col] = c[v] * alpha;
}

__device__ __forceinline__ void store_cfrag_bf16(bf16_t* p, int ld, int lane, v8f c, float alpha) {
  int col = lane & 15;
  int r0 = (lane >> 4) << 3;
#pragma unroll
  for (int v = 0; v < 8; ++v) p[(long)(r0 + v) * ld + col] = (bf16_t)(c[v] * alpha);
}

// ---- Tensor Data Mover: 1-D DMA of `bytes` (multiple of 4, here 8KB) global -> LDS ----
// D# group0: count=1 | lds_addr | global_addr[56:0] | type=2
// D# group1: data_size=2B, tensor_dim0 = tile_dim0 = nelem, dim0_stride = nelem
__device__ __forceinline__ void tdm_load_1d_to_lds(void* lds_dst, const void* gsrc, unsigned nelem2B) {
#if __has_builtin(__builtin_amdgcn_tensor_load_to_lds)
  unsigned lds = (unsigned)(unsigned long long)lds_dst;
  unsigned long long ga = (unsigned long long)gsrc;
  v4u g0;
  g0[0] = 1u;                                            // count=1, is_restore=0
  g0[1] = lds;                                           // lds_addr
  g0[2] = (unsigned)(ga & 0xffffffffull);                // global_addr[31:0]
  g0[3] = (unsigned)((ga >> 32) & 0x1ffffffull) | 0x80000000u; // addr[56:32] | type=2
  v8i g1;
  g1[0] = 0x10000;                   // workgroup_mask=0, data_size=1 (2 bytes)
  g1[1] = (int)(nelem2B << 16);      // tensor_dim0[15:0] << 16
  g1[2] = 1 << 16;                   // tensor_dim0[31:16]=0 | tensor_dim1=1
  g1[3] = (int)(nelem2B << 16);      // tile_dim0 << 16
  g1[4] = 0;                         // tile_dim1=0 (unused), tile_dim2=0
  g1[5] = (int)nelem2B;              // tensor_dim0_stride[31:0]
  g1[6] = 0;
  g1[7] = 0;
  v4i z4 = {};
#if defined(__clang_major__) && (__clang_major__ >= 23)
  v8i z8 = {};
  __builtin_amdgcn_tensor_load_to_lds(g0, g1, z4, z4, z8, 0);
#else
  __builtin_amdgcn_tensor_load_to_lds(g0, g1, z4, z4, 0);
#endif
#if __has_builtin(__builtin_amdgcn_s_wait_tensorcnt)
  __builtin_amdgcn_s_wait_tensorcnt(0);
#endif
#else
  // fallback: plain wave-cooperative copy (lane-strided b128)
  const uint4* s = (const uint4*)gsrc;
  uint4* d = (uint4*)lds_dst;
  int lane = (int)(threadIdx.x & 31);
  for (unsigned i = lane; i < (nelem2B >> 3); i += 32) d[i] = s[i];
#endif
}

// ---------------- weight conversion ----------------
__global__ void __launch_bounds__(256)
k_convert_weights(const float* __restrict__ tw, const float* __restrict__ pw,
                  const float* __restrict__ cw,
                  bf16_t* __restrict__ twb, bf16_t* __restrict__ pwb, bf16_t* __restrict__ cwb) {
  int i = blockIdx.x * 256 + threadIdx.x;
  if (i < DDH * CCH) { twb[i] = (bf16_t)tw[i]; pwb[i] = (bf16_t)pw[i]; }
  if (i < CCH * CCH) { cwb[i] = (bf16_t)cw[i]; }
}

// ---------------- x -> xT (hw-major) and xb (c-major), both bf16 ----------------
__global__ void __launch_bounds__(256)
k_convert_x(const float* __restrict__ x, bf16_t* __restrict__ xb, bf16_t* __restrict__ xT) {
  __shared__ float tile[32][33];
  int n = blockIdx.z;
  int a0 = blockIdx.x * 32;   // hw
  int c0 = blockIdx.y * 32;   // channel
  int tx = threadIdx.x & 31, ty = threadIdx.x >> 5;
  const float* xi = x + ((long)n * CCH + c0) * HWS + a0;
#pragma unroll
  for (int i = 0; i < 4; ++i) {
    int cc = ty + i * 8;
    float v = xi[(long)cc * HWS + tx];
    xb[((long)n * CCH + c0 + cc) * HWS + a0 + tx] = (bf16_t)v;
    tile[tx][cc] = v;            // tile[a][c]
  }
  __syncthreads();
#pragma unroll
  for (int i = 0; i < 4; ++i) {
    int aa = ty + i * 8;
    xT[((long)n * HWS + a0 + aa) * CCH + c0 + tx] = (bf16_t)tile[aa][tx];
  }
}

// ---------------- generic WMMA GEMM: C[M,N] = alpha * A[M,K] @ Bt[N,K]^T ----------------
// block tile 128(M) x 64(N), 8 waves: wave = 32x32
template<bool BF16_OUT>
__global__ void __launch_bounds__(256)
k_gemm_bt(const bf16_t* __restrict__ A, const bf16_t* __restrict__ Bt, void* __restrict__ Cv,
          int K, int N, long sA, long sB, long sC, float alpha) {
  int lane = threadIdx.x & 31, wid = threadIdx.x >> 5;
  int m0 = blockIdx.x * 128 + (wid & 3) * 32;
  int n0 = blockIdx.y * 64 + (wid >> 2) * 32;
  const bf16_t* Ab = A + (long)blockIdx.z * sA + (long)m0 * K;
  const bf16_t* Bb = Bt + (long)blockIdx.z * sB + (long)n0 * K;
  v8f acc[2][2] = {};
  for (int k = 0; k < K; k += 32) {
    __builtin_prefetch(Ab + k + 128, 0, 1);
    __builtin_prefetch(Bb + k + 128, 0, 1);
    v16bf a0 = load_frag(Ab + k, K, lane);
    v16bf a1 = load_frag(Ab + (long)16 * K + k, K, lane);
    v16bf b0 = load_frag(Bb + k, K, lane);
    v16bf b1 = load_frag(Bb + (long)16 * K + k, K, lane);
    acc[0][0] = wmma_bf(a0, b0, acc[0][0]);
    acc[0][1] = wmma_bf(a0, b1, acc[0][1]);
    acc[1][0] = wmma_bf(a1, b0, acc[1][0]);
    acc[1][1] = wmma_bf(a1, b1, acc[1][1]);
  }
#pragma unroll
  for (int mi = 0; mi < 2; ++mi)
#pragma unroll
    for (int ni = 0; ni < 2; ++ni) {
      long coff = (long)blockIdx.z * sC + (long)(m0 + mi * 16) * N + (n0 + ni * 16);
      if (BF16_OUT) store_cfrag_bf16((bf16_t*)Cv + coff, N, lane, acc[mi][ni], alpha);
      else          store_cfrag_f32((float*)Cv + coff, N, lane, acc[mi][ni], alpha);
    }
}

// ---------------- fused attention: scores -> softmax (LDS) -> value GEMM ----------------
// grid: (HWS/QT query tiles, NB images), 256 threads, scores tile QT x 1024 f32 in LDS
__device__ __forceinline__ v16bf afrag_from_lds(const float* sc, int m0, int k0, int lane) {
  const float* p = sc + (long)(m0 + (lane & 15)) * HWS + k0 + ((lane >> 4) << 3);
  float4 f0 = *(const float4*)(p);
  float4 f1 = *(const float4*)(p + 4);
  float4 f2 = *(const float4*)(p + 16);
  float4 f3 = *(const float4*)(p + 20);
  v16bf v;
  v[0]  = (bf16_t)f0.x; v[1]  = (bf16_t)f0.y; v[2]  = (bf16_t)f0.z; v[3]  = (bf16_t)f0.w;
  v[4]  = (bf16_t)f1.x; v[5]  = (bf16_t)f1.y; v[6]  = (bf16_t)f1.z; v[7]  = (bf16_t)f1.w;
  v[8]  = (bf16_t)f2.x; v[9]  = (bf16_t)f2.y; v[10] = (bf16_t)f2.z; v[11] = (bf16_t)f2.w;
  v[12] = (bf16_t)f3.x; v[13] = (bf16_t)f3.y; v[14] = (bf16_t)f3.z; v[15] = (bf16_t)f3.w;
  return v;
}

__global__ void __launch_bounds__(256)
k_attention(const bf16_t* __restrict__ thetaT, const bf16_t* __restrict__ phiT,
            const bf16_t* __restrict__ xb, bf16_t* __restrict__ fT) {
  __shared__ float sc[QT * HWS];        // 128 KB score/attn tile
  __shared__ bf16_t thA[QT * DDH];      // 8 KB theta A-tile (filled by TDM)
  __shared__ float red[256];
  __shared__ float rowm[QT];
  int img = blockIdx.y;
  int q0  = blockIdx.x * QT;
  int tid = threadIdx.x, lane = tid & 31, wid = tid >> 5;

  // DMA the theta tile for this query block into LDS (wave 0 issues; TDM ignores EXEC)
  const bf16_t* Ath0 = thetaT + ((long)img * HWS + q0) * DDH;
  if (wid == 0) tdm_load_1d_to_lds(thA, Ath0, QT * DDH);
  __syncthreads();

  // Phase 1: S[q, b] = (1/16) * theta[q,:] . phi[b,:]   (M=QT, N=1024, K=128)
  {
    const bf16_t* Bph = phiT + (long)img * HWS * DDH;
    int nbase = wid * 128;               // each wave: 128 key columns
    for (int nc = 0; nc < 128; nc += 16) {
      int n0 = nbase + nc;
      v8f c0 = {}, c1 = {};
      for (int k = 0; k < DDH; k += 32) {
        v16bf a0 = load_frag(thA + k, DDH, lane);            // ds_load_b128 from LDS
        v16bf a1 = load_frag(thA + 16 * DDH + k, DDH, lane);
        v16bf b  = load_frag(Bph + (long)n0 * DDH + k, DDH, lane);
        c0 = wmma_bf(a0, b, c0);
        c1 = wmma_bf(a1, b, c1);
      }
      store_cfrag_f32(sc + n0, HWS, lane, c0, 0.0625f);
      store_cfrag_f32(sc + 16 * HWS + n0, HWS, lane, c1, 0.0625f);
    }
  }
  __syncthreads();

  // Phase 2: row softmax over 1024 keys, normalized attn written back in-place (f32)
  {
    int row = tid >> 3, part = tid & 7;
    float4* rp = (float4*)(sc + (long)row * HWS + part * 128);
    float m = -3.4e38f;
#pragma unroll 4
    for (int i = 0; i < 32; ++i) {
      float4 f = rp[i];
      m = fmaxf(m, fmaxf(fmaxf(f.x, f.y), fmaxf(f.z, f.w)));
    }
    red[tid] = m;
    __syncthreads();
    if (part == 0) {
      float mm = red[tid];
#pragma unroll
      for (int i = 1; i < 8; ++i) mm = fmaxf(mm, red[tid + i]);
      rowm[row] = mm;
    }
    __syncthreads();
    float mm = rowm[row];
    float s = 0.f;
#pragma unroll 4
    for (int i = 0; i < 32; ++i) {
      float4 f = rp[i];
      f.x = __expf(f.x - mm); f.y = __expf(f.y - mm);
      f.z = __expf(f.z - mm); f.w = __expf(f.w - mm);
      s += f.x + f.y + f.z + f.w;
      rp[i] = f;
    }
    red[tid] = s;
    __syncthreads();
    if (part == 0) {
      float ss = 0.f;
#pragma unroll
      for (int i = 0; i < 8; ++i) ss += red[tid + i];
      rowm[row] = 1.f / ss;
    }
    __syncthreads();
    float inv = rowm[row];
#pragma unroll 4
    for (int i = 0; i < 32; ++i) {
      float4 f = rp[i];
      f.x *= inv; f.y *= inv; f.z *= inv; f.w *= inv;
      rp[i] = f;
    }
  }
  __syncthreads();

  // Phase 3: fT[q, i] = attn[q, :] . x[i, :]   (M=QT, N=256, K=1024; Bt = xb [c, hw])
  {
    const bf16_t* Bx = xb + (long)img * CCH * HWS;
    int n0 = wid * 32;                  // each wave: 32 output channels
    v8f acc[2][2] = {};
    for (int k = 0; k < HWS; k += 32) {
      v16bf a0 = afrag_from_lds(sc, 0, k, lane);
      v16bf a1 = afrag_from_lds(sc, 16, k, lane);
      v16bf b0 = load_frag(Bx + (long)n0 * HWS + k, HWS, lane);
      v16bf b1 = load_frag(Bx + (long)(n0 + 16) * HWS + k, HWS, lane);
      acc[0][0] = wmma_bf(a0, b0, acc[0][0]);
      acc[0][1] = wmma_bf(a0, b1, acc[0][1]);
      acc[1][0] = wmma_bf(a1, b0, acc[1][0]);
      acc[1][1] = wmma_bf(a1, b1, acc[1][1]);
    }
    bf16_t* O = fT + ((long)img * HWS + q0) * CCH;
#pragma unroll
    for (int mi = 0; mi < 2; ++mi)
#pragma unroll
      for (int ni = 0; ni < 2; ++ni)
        store_cfrag_bf16(O + (long)(mi * 16) * CCH + n0 + ni * 16, CCH, lane, acc[mi][ni], 1.f);
  }
}

// ---------------- BN stats: deterministic two-stage reduction ----------------
__global__ void __launch_bounds__(256)
k_stats_partial(const float* __restrict__ outT, float* __restrict__ psum, float* __restrict__ psq) {
  long base = (long)blockIdx.x * 8192;   // 32 rows of 256 channels
  int d = threadIdx.x;
  float s = 0.f, q = 0.f;
#pragma unroll 4
  for (int i = 0; i < 32; ++i) {
    float v = outT[base + (long)i * 256 + d];
    s += v; q += v * v;
  }
  psum[(long)blockIdx.x * 256 + d] = s;
  psq [(long)blockIdx.x * 256 + d] = q;
}

__global__ void __launch_bounds__(256)
k_stats_final(const float* __restrict__ psum, const float* __restrict__ psq, float* __restrict__ stats) {
  int d = threadIdx.x;
  float s = 0.f, q = 0.f;
  for (int j = 0; j < 1024; ++j) { s += psum[(long)j * 256 + d]; q += psq[(long)j * 256 + d]; }
  float mean = s * (1.f / 32768.f);
  float var  = q * (1.f / 32768.f) - mean * mean;
  stats[d]       = mean;
  stats[256 + d] = rsqrtf(var + 1e-5f);
}

// ---------------- BN + residual + [hw,c]->[c,hw] transpose epilogue ----------------
__global__ void __launch_bounds__(256)
k_finalize(const float* __restrict__ outT, const float* __restrict__ x,
           const float* __restrict__ stats, const float* __restrict__ gamma,
           const float* __restrict__ beta, float* __restrict__ out) {
  __shared__ float tile[32][33];
  int n = blockIdx.z;
  int a0 = blockIdx.x * 32;   // hw
  int d0 = blockIdx.y * 32;   // channel
  int tx = threadIdx.x & 31, ty = threadIdx.x >> 5;
  const float* src = outT + ((long)n * HWS + a0) * CCH + d0;
#pragma unroll
  for (int i = 0; i < 4; ++i) {
    int aa = ty + i * 8;
    tile[aa][tx] = src[(long)aa * CCH + tx];   // tile[a][c]
  }
  __syncthreads();
#pragma unroll
  for (int i = 0; i < 4; ++i) {
    int dd = ty + i * 8;
    int d = d0 + dd;
    float mean = stats[d], rstd = stats[256 + d];
    float g = gamma[d], b = beta[d];
    long oi = (((long)n * CCH + d) * HWS) + a0 + tx;
    out[oi] = (tile[tx][dd] - mean) * rstd * g + b + x[oi];
  }
}

extern "C" void kernel_launch(void* const* d_in, const int* in_sizes, int n_in,
                              void* d_out, int out_size, void* d_ws, size_t ws_size,
                              hipStream_t stream) {
  const float* x       = (const float*)d_in[0];
  const float* theta_w = (const float*)d_in[1];
  const float* phi_w   = (const float*)d_in[2];
  const float* conv_w  = (const float*)d_in[3];
  const float* gamma   = (const float*)d_in[4];
  const float* beta    = (const float*)d_in[5];
  float* out = (float*)d_out;

  char* ws = (char*)d_ws;
  size_t off = 0;
  auto alloc = [&](size_t bytes) -> void* {
    void* p = ws + off;
    off += (bytes + 255) & ~(size_t)255;
    return p;
  };
  bf16_t* XT   = (bf16_t*)alloc((size_t)NB * HWS * CCH * 2);  // x^T  [n][hw][c]
  bf16_t* XB   = (bf16_t*)alloc((size_t)NB * CCH * HWS * 2);  // x    [n][c][hw] bf16
  bf16_t* THT  = (bf16_t*)alloc((size_t)NB * HWS * DDH * 2);  // theta^T [n][hw][d]
  bf16_t* PHT  = (bf16_t*)alloc((size_t)NB * HWS * DDH * 2);  // phi^T   [n][hw][d]
  bf16_t* FT   = (bf16_t*)alloc((size_t)NB * HWS * CCH * 2);  // f^T  [n][hw][c]
  float*  OUTT = (float*) alloc((size_t)NB * HWS * CCH * 4);  // conv out [n][hw][c]
  bf16_t* WT   = (bf16_t*)alloc((size_t)DDH * CCH * 2);
  bf16_t* WP   = (bf16_t*)alloc((size_t)DDH * CCH * 2);
  bf16_t* WC   = (bf16_t*)alloc((size_t)CCH * CCH * 2);
  float*  PSUM = (float*) alloc((size_t)1024 * 256 * 4);
  float*  PSQ  = (float*) alloc((size_t)1024 * 256 * 4);
  float*  STAT = (float*) alloc((size_t)512 * 4);

  k_convert_weights<<<256, 256, 0, stream>>>(theta_w, phi_w, conv_w, WT, WP, WC);
  k_convert_x<<<dim3(HWS / 32, CCH / 32, NB), 256, 0, stream>>>(x, XB, XT);

  // theta^T = x^T @ theta_w^T, phi^T = x^T @ phi_w^T   (M=1024, N=128, K=256)
  k_gemm_bt<true><<<dim3(8, 2, NB), 256, 0, stream>>>(
      XT, WT, THT, CCH, DDH, (long)HWS * CCH, 0, (long)HWS * DDH, 1.f);
  k_gemm_bt<true><<<dim3(8, 2, NB), 256, 0, stream>>>(
      XT, WP, PHT, CCH, DDH, (long)HWS * CCH, 0, (long)HWS * DDH, 1.f);

  // fused scores -> softmax -> value matmul
  k_attention<<<dim3(HWS / QT, NB), 256, 0, stream>>>(THT, PHT, XB, FT);

  // out^T = f^T @ conv_w^T   (M=1024, N=256, K=256), f32 out
  k_gemm_bt<false><<<dim3(8, 4, NB), 256, 0, stream>>>(
      FT, WC, OUTT, CCH, CCH, (long)HWS * CCH, 0, (long)HWS * CCH, 1.f);

  k_stats_partial<<<1024, 256, 0, stream>>>(OUTT, PSUM, PSQ);
  k_stats_final<<<1, 256, 0, stream>>>(PSUM, PSQ, STAT);
  k_finalize<<<dim3(HWS / 32, CCH / 32, NB), 256, 0, stream>>>(OUTT, x, STAT, gamma, beta, out);

  (void)in_sizes; (void)n_in; (void)out_size; (void)ws_size;
}
